// Decoder_82643760710187
// MI455X (gfx1250) — compile-verified, matrix operand order checked by reading, not run on hardware
//
#include <hip/hip_runtime.h>

typedef __attribute__((ext_vector_type(16))) __bf16 v16bf;
typedef __attribute__((ext_vector_type(8)))  float  v8f;

#define N_CHROM 23
#define BINS 5000
#define BATCH 16
#define SAMP 512
#define PRED_ELEMS (N_CHROM * BATCH * SAMP * 3)   // 565248
#define TGT_ELEMS  (N_CHROM * BATCH * SAMP)       // 188416

__device__ __forceinline__ unsigned short f2bf(float f) {
    unsigned int u = __builtin_bit_cast(unsigned int, f);
    u += 0x7FFFu + ((u >> 16) & 1u);              // round-to-nearest-even
    return (unsigned short)(u >> 16);
}

// ---------------------------------------------------------------------------
// P1: Wcomb[k,n] = sum_d W_bin[d,k] * W_fc[n,d], packed directly into the
// CDNA5 16-bit B-matrix (32x16) fragment layout, bf16, columns n>=3 zeroed.
// frag is uint16[8 chunks][32 lanes][16 elems] = 8KB.
// Element map (mirrors A-layout from ISA 7.12.2): within K-chunk ch,
//   r = k % 32:  r in [0,8)  -> lanes 0-15,  e = r
//                r in [8,16) -> lanes 16-31, e = r-8
//                r in [16,24)-> lanes 0-15,  e = r-8
//                r in [24,32)-> lanes 16-31, e = r-16
// ---------------------------------------------------------------------------
__global__ __launch_bounds__(256) void prep_wcomb(const float* __restrict__ W_bin,
                                                  const float* __restrict__ W_fc,
                                                  unsigned short* __restrict__ frag) {
    const int k = threadIdx.x;                    // 0..255
    float wc0 = 0.f, wc1 = 0.f, wc2 = 0.f;
    for (int d = 0; d < 512; ++d) {
        float wb = W_bin[d * 256 + k];            // coalesced across k
        wc0 += wb * W_fc[d];
        wc1 += wb * W_fc[1024 + d];
        wc2 += wb * W_fc[2048 + d];
    }
    const int ch = k >> 5;
    const int r  = k & 31;
    const int hi = (r >> 3) & 1;
    const int e  = (r & 7) | ((r >> 4) << 3);
    const unsigned short v0 = f2bf(wc0), v1 = f2bf(wc1), v2 = f2bf(wc2);
    for (int l = 0; l < 16; ++l) {
        unsigned short v = (l == 0) ? v0 : (l == 1) ? v1 : (l == 2) ? v2 : (unsigned short)0;
        frag[(ch * 32 + l + hi * 16) * 16 + e] = v;
    }
}

// ---------------------------------------------------------------------------
// P2: Weos_comb[e,n] = sum_d W_eos[d,e] * W_fc[n,512+d]   (512 x 3)
//     bconst[n]      = b_fc[n] + sum_d b_eos[d] * W_fc[n,512+d]
// ---------------------------------------------------------------------------
__global__ __launch_bounds__(256) void prep_weos(const float* __restrict__ W_eos,
                                                 const float* __restrict__ W_fc,
                                                 const float* __restrict__ b_eos,
                                                 const float* __restrict__ b_fc,
                                                 float* __restrict__ weos_comb,
                                                 float* __restrict__ bconst) {
    const int e = blockIdx.x * 256 + threadIdx.x; // 0..511
    float a0 = 0.f, a1 = 0.f, a2 = 0.f;
    for (int d = 0; d < 512; ++d) {
        float w = W_eos[d * 512 + e];             // coalesced across e
        a0 += w * W_fc[512 + d];
        a1 += w * W_fc[1024 + 512 + d];
        a2 += w * W_fc[2048 + 512 + d];
    }
    weos_comb[e * 3 + 0] = a0;
    weos_comb[e * 3 + 1] = a1;
    weos_comb[e * 3 + 2] = a2;
    if (e < 3) {
        float s = b_fc[e];
        for (int d = 0; d < 512; ++d) s += b_eos[d] * W_fc[e * 1024 + 512 + d];
        bconst[e] = s;
    }
}

// ---------------------------------------------------------------------------
// P3: econst[c,b,n] = sum_e eos_emb[b,c,e] * Weos_comb[e,n] + bconst[n]
// one block per (c,b)
// ---------------------------------------------------------------------------
__global__ __launch_bounds__(128) void prep_econst(const float* __restrict__ eos_emb,
                                                   const float* __restrict__ weos_comb,
                                                   const float* __restrict__ bconst,
                                                   float* __restrict__ econst) {
    const int cb = blockIdx.x;                    // c*16 + b
    const int c = cb >> 4, b = cb & 15;
    const float* x = eos_emb + (b * N_CHROM + c) * 512;
    __shared__ float r0[128], r1[128], r2[128];
    const int t = threadIdx.x;
    float a0 = 0.f, a1 = 0.f, a2 = 0.f;
    for (int e = t; e < 512; e += 128) {
        float v = x[e];
        a0 += v * weos_comb[e * 3 + 0];
        a1 += v * weos_comb[e * 3 + 1];
        a2 += v * weos_comb[e * 3 + 2];
    }
    r0[t] = a0; r1[t] = a1; r2[t] = a2;
    __syncthreads();
    for (int s = 64; s > 0; s >>= 1) {
        if (t < s) { r0[t] += r0[t + s]; r1[t] += r1[t + s]; r2[t] += r2[t + s]; }
        __syncthreads();
    }
    if (t == 0) {
        econst[cb * 3 + 0] = r0[0] + bconst[0];
        econst[cb * 3 + 1] = r1[0] + bconst[1];
        econst[cb * 3 + 2] = r2[0] + bconst[2];
    }
}

// ---------------------------------------------------------------------------
// Main fused kernel: gather 16 embedding rows per wave, contract against
// packed Wcomb via 8x v_wmma_f32_16x16x32_bf16 (K=256), add econst, ReLU.
// HBM-gather bound: ~193 MB total -> ~8.3 us roofline at 23.3 TB/s.
// ---------------------------------------------------------------------------
__global__ __launch_bounds__(256) void fused_pred(const int* __restrict__ sampled_indices,
                                                  const float* __restrict__ table,
                                                  const unsigned short* __restrict__ frag,
                                                  const float* __restrict__ econst,
                                                  float* __restrict__ pred) {
    const int wave = threadIdx.x >> 5;
    const int lane = threadIdx.x & 31;
    const int tile = blockIdx.x * 8 + wave;       // 0..11775
    const int c    = tile >> 9;                   // / (16 b * 32 s-tiles)
    const int rem  = tile & 511;
    const int b    = rem >> 5;
    const int s0   = (rem & 31) << 4;
    const int m    = lane & 15;                   // row within tile / output col
    const int hi   = lane >> 4;
    const int cb   = c * 16 + b;

    const long long vidx = (long long)sampled_indices[cb * SAMP + s0 + m] + (long long)c * BINS;
    const float* row = table + vidx * 256;        // 1 KB embedding row
    const uint4* frags = (const uint4*)frag;

    v8f acc = {0.f, 0.f, 0.f, 0.f, 0.f, 0.f, 0.f, 0.f};

#pragma unroll
    for (int ch = 0; ch < 8; ++ch) {
        // A fragment per ISA 7.12.2 (16-bit A 16x32): lane<16 holds K=ch*32+{0..7,16..23},
        // lane>=16 holds K=ch*32+{8..15,24..31}; lane pair (L, L+16) covers the full row.
        const float4* p0 = (const float4*)(row + ch * 32 + hi * 8);
        const float4* p1 = (const float4*)(row + ch * 32 + 16 + hi * 8);
        const float4 f0 = p0[0], f1 = p0[1];
        const float4 g0 = p1[0], g1 = p1[1];

        const uint4 q0 = frags[(ch * 32 + lane) * 2 + 0];
        const uint4 q1 = frags[(ch * 32 + lane) * 2 + 1];

        v16bf a;
        a[0]  = (__bf16)f0.x; a[1]  = (__bf16)f0.y; a[2]  = (__bf16)f0.z; a[3]  = (__bf16)f0.w;
        a[4]  = (__bf16)f1.x; a[5]  = (__bf16)f1.y; a[6]  = (__bf16)f1.z; a[7]  = (__bf16)f1.w;
        a[8]  = (__bf16)g0.x; a[9]  = (__bf16)g0.y; a[10] = (__bf16)g0.z; a[11] = (__bf16)g0.w;
        a[12] = (__bf16)g1.x; a[13] = (__bf16)g1.y; a[14] = (__bf16)g1.z; a[15] = (__bf16)g1.w;

        union { uint4 q[2]; v16bf v; } bu;
        bu.q[0] = q0; bu.q[1] = q1;

        acc = __builtin_amdgcn_wmma_f32_16x16x32_bf16(false, a, false, bu.v,
                                                      (short)0, acc, false, false);
    }

    // D layout: lane n (n<16): VGPR j = D[M=j][N=n]; lane n+16: D[M=j+8][N=n].
    if (m < 3) {
        const float e = econst[cb * 3 + m];
        int base = (cb * SAMP + s0 + hi * 8) * 3 + m;
#pragma unroll
        for (int j = 0; j < 8; ++j) {
            float v = acc[j] + e;
            pred[base + j * 3] = v > 0.f ? v : 0.f;
        }
    }
}

__global__ __launch_bounds__(256) void copy_targets(const int* __restrict__ t,
                                                    int* __restrict__ out, int n) {
    int i = blockIdx.x * 256 + threadIdx.x;
    if (i < n) out[i] = t[i];
}

extern "C" void kernel_launch(void* const* d_in, const int* in_sizes, int n_in,
                              void* d_out, int out_size, void* d_ws, size_t ws_size,
                              hipStream_t stream) {
    const float* eos_emb         = (const float*)d_in[0];
    const int*   sampled_indices = (const int*)d_in[1];
    const int*   targets         = (const int*)d_in[2];
    const float* table           = (const float*)d_in[3];
    const float* W_bin           = (const float*)d_in[4];
    const float* W_eos           = (const float*)d_in[5];
    const float* b_eos           = (const float*)d_in[6];
    const float* W_fc            = (const float*)d_in[7];
    const float* b_fc            = (const float*)d_in[8];

    char* ws = (char*)d_ws;
    unsigned short* frag      = (unsigned short*)ws;              // 8 KB
    float*          weos_comb = (float*)(ws + 8192);              // 6 KB
    float*          bconst    = (float*)(ws + 8192 + 6144);       // 12 B (pad 64)
    float*          econst    = (float*)(ws + 8192 + 6144 + 64);  // 4.4 KB

    prep_wcomb <<<1,   256, 0, stream>>>(W_bin, W_fc, frag);
    prep_weos  <<<2,   256, 0, stream>>>(W_eos, W_fc, b_eos, b_fc, weos_comb, bconst);
    prep_econst<<<368, 128, 0, stream>>>(eos_emb, weos_comb, bconst, econst);

    float* pred = (float*)d_out;
    fused_pred <<<1472, 256, 0, stream>>>(sampled_indices, table, frag, econst, pred);

    int* tout = (int*)d_out + PRED_ELEMS;
    copy_targets<<<(TGT_ELEMS + 255) / 256, 256, 0, stream>>>(targets, tout, TGT_ELEMS);
}
